// CustomLSTM_24352464569687
// MI455X (gfx1250) — matrix operand order repaired and checked by reading, not compiled, not run on hardware
//
#include <hip/hip_runtime.h>

// ---------------------------------------------------------------------------
// CustomLSTM for MI455X (gfx1250, wave32, WMMA)
//   x[B=64,S=512,I=512] f32, W[512,4096] f32, U[1024,4096] f32, bias[4096] f32
//   out = hidden_seq[B,S,H=1024] f32  ++  h_last[B,H] ++ c_last[B,H]
// ---------------------------------------------------------------------------

#define ISZ   512
#define HS    1024
#define G4    4096     // 4*HS
#define BATCH 64
#define SEQL  512
#define NWG   64       // persistent workgroups, one per 16 h-columns
#define UTP   1032     // LDS row pitch for Ut slice (elements): 516 dwords,
                       // 516 % 64 == 4 -> lane rows sweep all 64 banks

// dynamic LDS: gates[4][16][64] f32 + cell[16][64] f32 + ut[64][UTP] bf16
#define SMEM_GATES_ELEMS (4 * 16 * BATCH)
#define SMEM_CELL_ELEMS  (16 * BATCH)
#define SMEM_BYTES ((SMEM_GATES_ELEMS + SMEM_CELL_ELEMS) * 4 + 64 * UTP * 2)

typedef __attribute__((ext_vector_type(16))) __bf16 v16bf;
typedef __attribute__((ext_vector_type(8)))  __bf16 v8bf;
typedef __attribute__((ext_vector_type(8)))  float  v8f;

__device__ __forceinline__ v8f wmma_bf16(v16bf a, v16bf b, v8f c) {
    // D = A(16x32 bf16) * B(32x16 bf16) + C(16x16 f32)
    return __builtin_amdgcn_wmma_f32_16x16x32_bf16(
        false, a, false, b, (short)0, c, false, false);
}

// A-matrix (16x32 bf16) lane load, contiguous K order matching B:
// lanes 0-15: K in [kb, kb+16); lanes 16-31: K in [kb+16, kb+32).
__device__ __forceinline__ v16bf load_a_tile(const __bf16* __restrict__ row,
                                             int kb, int hlf) {
    v8bf lo = *(const v8bf*)(row + kb + hlf * 16);
    v8bf hi = *(const v8bf*)(row + kb + hlf * 16 + 8);
    v16bf a;
#pragma unroll
    for (int j = 0; j < 8; ++j) { a[j] = lo[j]; a[8 + j] = hi[j]; }
    return a;
}

// B-matrix (32x16 bf16): per-lane 32 contiguous bytes at row[kb + 16*half],
// fetched as two 16B chunks (LDS rows are 16B- but not 32B-aligned).
__device__ __forceinline__ v16bf load_b_lds(const __bf16* row, int off) {
    v8bf lo = *(const v8bf*)(row + off);
    v8bf hi = *(const v8bf*)(row + off + 8);
    v16bf b;
#pragma unroll
    for (int j = 0; j < 8; ++j) { b[j] = lo[j]; b[8 + j] = hi[j]; }
    return b;
}

__device__ __forceinline__ float sigmoidf_(float x) {
    return 1.0f / (1.0f + __expf(-x));
}

// -------------------------- prep kernels -----------------------------------

__global__ void cvt_f32_to_bf16(const float* __restrict__ src,
                                __bf16* __restrict__ dst, int n) {
    int i = blockIdx.x * blockDim.x + threadIdx.x;
    if (i < n) dst[i] = (__bf16)src[i];
}

// src[R][C] f32  ->  dst[C][R] bf16   (transpose + convert)
__global__ void transpose_to_bf16(const float* __restrict__ src,
                                  __bf16* __restrict__ dst, int R, int C) {
    int i = blockIdx.x * blockDim.x + threadIdx.x;
    if (i < R * C) {
        int r = i / C, c = i % C;
        dst[(size_t)c * R + r] = (__bf16)src[i];
    }
}

// ------------------- phase 1: xW[s][g][b] = x@W + bias ---------------------
// grid = (SEQL, G4/128); block = 256 (8 waves). Each wave: one 16-row g tile,
// 4 batch tiles of 16, K = ISZ.
__global__ __launch_bounds__(256) void proj_kernel(
    const __bf16* __restrict__ xb,   // [B][S][I] bf16
    const __bf16* __restrict__ Wt,   // [G4][I]   bf16 (W transposed)
    const float*  __restrict__ bias, // [G4]
    float*        __restrict__ xW)   // [S][G4][B] f32
{
    const int s    = blockIdx.x;
    const int gb   = blockIdx.y;           // block of 128 g-rows
    const int tid  = threadIdx.x;
    const int lane = tid & 31;
    const int wave = tid >> 5;
    const int hlf  = lane >> 4;
    const int l15  = lane & 15;
    const int gbase = gb * 128 + wave * 16;     // this wave's 16 g-rows

    const __bf16* arow = Wt + (size_t)(gbase + l15) * ISZ;
    const __bf16* xrow[4];
#pragma unroll
    for (int nt = 0; nt < 4; ++nt)
        xrow[nt] = xb + ((size_t)(nt * 16 + l15) * SEQL + s) * ISZ;

    v8f acc[4] = {};
    for (int kb = 0; kb < ISZ; kb += 32) {
        v16bf a = load_a_tile(arow, kb, hlf);
#pragma unroll
        for (int nt = 0; nt < 4; ++nt) {
            v16bf b = *(const v16bf*)(xrow[nt] + kb + hlf * 16);
            acc[nt] = wmma_bf16(a, b, acc[nt]);
        }
    }

    // bias: rows gbase + hlf*8 + i
    float bb[8];
#pragma unroll
    for (int i = 0; i < 8; ++i) bb[i] = bias[gbase + hlf * 8 + i];

    float* obase = xW + (size_t)s * G4 * BATCH;
#pragma unroll
    for (int nt = 0; nt < 4; ++nt) {
#pragma unroll
        for (int i = 0; i < 8; ++i) {
            int grow = gbase + hlf * 8 + i;
            obase[(size_t)grow * BATCH + nt * 16 + l15] = acc[nt][i] + bb[i];
        }
    }
}

// ------------------- phase 2: persistent recurrent scan --------------------

__device__ __forceinline__ void grid_barrier(unsigned* bar, unsigned target) {
    __syncthreads();
    if (threadIdx.x == 0) {
        __hip_atomic_fetch_add(bar, 1u, __ATOMIC_ACQ_REL,
                               __HIP_MEMORY_SCOPE_AGENT);
        while (__hip_atomic_load(bar, __ATOMIC_ACQUIRE,
                                 __HIP_MEMORY_SCOPE_AGENT) < target) {
            __builtin_amdgcn_s_sleep(2);
        }
    }
    __syncthreads();
}

__global__ __launch_bounds__(256) void lstm_scan_kernel(
    const float*  __restrict__ xW,   // [S][G4][B]
    const __bf16* __restrict__ Ut,   // [G4][HS]  (U transposed)
    __bf16*       __restrict__ hbuf, // [2][B][HS] double buffered
    float*        __restrict__ out,  // hidden_seq ++ h_last ++ c_last
    unsigned*     __restrict__ bar)
{
    extern __shared__ char smem[];
    float*  gates = (float*)smem;                       // [4][16][64]
    float*  cl    = gates + SMEM_GATES_ELEMS;           // [16][64]
    __bf16* utl   = (__bf16*)(cl + SMEM_CELL_ELEMS);    // [64][UTP]

    const int wg   = blockIdx.x;      // 0..63
    const int c0   = wg * 16;         // h-column base owned by this WG
    const int tid  = threadIdx.x;
    const int lane = tid & 31;
    const int wave = tid >> 5;        // 0..7
    const int Mt   = wave & 3;        // batch tile (16 rows)
    const int g0   = (wave >> 2) * 2; // gates g0, g0+1
    const int hlf  = lane >> 4;
    const int l15  = lane & 15;

    // stage this WG's Ut slice into LDS: 64 rows (gate*16+n) x 1024 K, bf16.
    // 16B segments: 64 rows * 128 segs; LDS row pitch UTP keeps 16B alignment
    // (UTP*2 = 2064 = 129*16) and conflict-free lane access.
    for (int idx = tid; idx < 64 * 128; idx += 256) {
        int lrow = idx >> 7;          // 0..63
        int seg  = idx & 127;         // 16B segment within row
        int g    = lrow >> 4;
        int n    = lrow & 15;
        const uint4* src =
            (const uint4*)(Ut + (size_t)(g * HS + c0 + n) * HS) + seg;
        *(uint4*)((char*)(utl + (size_t)lrow * UTP) + seg * 16) = *src;
    }
    for (int e = tid; e < 16 * BATCH; e += 256) cl[e] = 0.0f;
    __syncthreads();

    const __bf16* brow0 = utl + (size_t)((g0    ) * 16 + l15) * UTP;
    const __bf16* brow1 = utl + (size_t)((g0 + 1) * 16 + l15) * UTP;
    const size_t  HSEQ  = (size_t)BATCH * SEQL * HS;

#pragma unroll 1
    for (int s = 0; s < SEQL; ++s) {
        const __bf16* hrd = hbuf + (size_t)(s & 1) * BATCH * HS;
        __bf16*       hwr = hbuf + (size_t)((s + 1) & 1) * BATCH * HS;
        const __bf16* arow = hrd + (size_t)(Mt * 16 + l15) * HS;

        // seed accumulators with xW[s][gcol][b] (issues f32 loads up front,
        // overlapping their latency with the first WMMAs)
        v8f acc0, acc1;
        {
            const float* base = xW + (size_t)s * G4 * BATCH;
            const float* p0 = base + (size_t)(g0 * HS + c0 + l15) * BATCH
                                   + Mt * 16 + hlf * 8;
            const float* p1 = base + (size_t)((g0 + 1) * HS + c0 + l15) * BATCH
                                   + Mt * 16 + hlf * 8;
            float4 q0 = *(const float4*)p0, q1 = *(const float4*)(p0 + 4);
            float4 r0 = *(const float4*)p1, r1 = *(const float4*)(p1 + 4);
            acc0[0] = q0.x; acc0[1] = q0.y; acc0[2] = q0.z; acc0[3] = q0.w;
            acc0[4] = q1.x; acc0[5] = q1.y; acc0[6] = q1.z; acc0[7] = q1.w;
            acc1[0] = r0.x; acc1[1] = r0.y; acc1[2] = r0.z; acc1[3] = r0.w;
            acc1[4] = r1.x; acc1[5] = r1.y; acc1[6] = r1.z; acc1[7] = r1.w;
        }

#pragma unroll 4
        for (int kb = 0; kb < HS; kb += 32) {
            v16bf a  = load_a_tile(arow, kb, hlf);
            v16bf b0 = load_b_lds(brow0, kb + hlf * 16);
            v16bf b1 = load_b_lds(brow1, kb + hlf * 16);
            acc0 = wmma_bf16(a, b0, acc0);
            acc1 = wmma_bf16(a, b1, acc1);
        }

        // stage gates to LDS: gates[g][col=l15][b = Mt*16 + hlf*8 + i]
        {
            float4* d0 = (float4*)(gates + ((g0    ) * 16 + l15) * BATCH
                                         + Mt * 16 + hlf * 8);
            float4* d1 = (float4*)(gates + ((g0 + 1) * 16 + l15) * BATCH
                                         + Mt * 16 + hlf * 8);
            d0[0] = make_float4(acc0[0], acc0[1], acc0[2], acc0[3]);
            d0[1] = make_float4(acc0[4], acc0[5], acc0[6], acc0[7]);
            d1[0] = make_float4(acc1[0], acc1[1], acc1[2], acc1[3]);
            d1[1] = make_float4(acc1[4], acc1[5], acc1[6], acc1[7]);
        }
        __syncthreads();

        // elementwise gate math + state update for this WG's 16 columns
        for (int e = tid; e < 16 * BATCH; e += 256) {
            int col = e >> 6, b = e & 63;
            float ig = sigmoidf_(gates[(0 * 16 + col) * BATCH + b]);
            float fg = sigmoidf_(gates[(1 * 16 + col) * BATCH + b]);
            float gg = sigmoidf_(gates[(2 * 16 + col) * BATCH + b]);  // ref: sigmoid
            float og = sigmoidf_(gates[(3 * 16 + col) * BATCH + b]);
            float cn = fg * cl[e] + ig * gg;
            float hn = og * tanhf(cn);
            cl[e] = cn;
            hwr[(size_t)b * HS + c0 + col] = (__bf16)hn;
            out[((size_t)b * SEQL + s) * HS + c0 + col] = hn;
            if (s == SEQL - 1) {
                out[HSEQ + (size_t)b * HS + c0 + col] = hn;                      // h_last
                out[HSEQ + (size_t)BATCH * HS + (size_t)b * HS + c0 + col] = cn; // c_last
            }
        }

        grid_barrier(bar, (unsigned)((s + 1) * NWG));
    }
}

// ---------------------------------------------------------------------------

extern "C" void kernel_launch(void* const* d_in, const int* in_sizes, int n_in,
                              void* d_out, int out_size, void* d_ws, size_t ws_size,
                              hipStream_t stream) {
    const float* x    = (const float*)d_in[0];   // [64,512,512]
    const float* W    = (const float*)d_in[1];   // [512,4096]
    const float* U    = (const float*)d_in[2];   // [1024,4096]
    const float* bias = (const float*)d_in[3];   // [4096]
    float* out = (float*)d_out;

    // workspace carve-up (256B aligned regions)
    char* p = (char*)d_ws;
    float*  xW = (float*)p;  p += (size_t)SEQL * G4 * BATCH * sizeof(float);   // 512 MB
    __bf16* xb = (__bf16*)p; p += (size_t)BATCH * SEQL * ISZ * 2;              // 32 MB
    __bf16* Wt = (__bf16*)p; p += (size_t)G4 * ISZ * 2;                        // 4 MB
    __bf16* Ut = (__bf16*)p; p += (size_t)G4 * HS * 2;                         // 8 MB
    __bf16* hb = (__bf16*)p; p += (size_t)2 * BATCH * HS * 2;                  // 256 KB
    unsigned* bar = (unsigned*)p;

    // allow the large dynamic-LDS allocation for the scan kernel (~149 KB of
    // the WGP's 320 KB); harmless no-op if already set.
    hipFuncSetAttribute((const void*)lstm_scan_kernel,
                        hipFuncAttributeMaxDynamicSharedMemorySize, SMEM_BYTES);

    // deterministic per-call init (graph-capture-legal memset nodes)
    hipMemsetAsync(hb, 0, (size_t)2 * BATCH * HS * 2, stream);
    hipMemsetAsync(bar, 0, 256, stream);

    // prep: bf16 conversions / transposes
    {
        int n = BATCH * SEQL * ISZ;
        cvt_f32_to_bf16<<<(n + 255) / 256, 256, 0, stream>>>(x, xb, n);
    }
    {
        int n = ISZ * G4;
        transpose_to_bf16<<<(n + 255) / 256, 256, 0, stream>>>(W, Wt, ISZ, G4);
    }
    {
        int n = HS * G4;
        transpose_to_bf16<<<(n + 255) / 256, 256, 0, stream>>>(U, Ut, HS, G4);
    }

    // phase 1: batched input projection (+bias) via WMMA
    proj_kernel<<<dim3(SEQL, G4 / 128), 256, 0, stream>>>(xb, Wt, bias, xW);

    // phase 2: persistent recurrent scan, grid-wide barrier per timestep
    lstm_scan_kernel<<<NWG, 256, SMEM_BYTES, stream>>>(xW, Ut, hb, out, bar);
}
